// Causal_Aware_Attention_71863392796909
// MI455X (gfx1250) — matrix operand order repaired and checked by reading, not compiled, not run on hardware
//
#include <hip/hip_runtime.h>

// ---- problem constants (match reference) ----
#define D_   512
#define H_   8
#define HD_  64
#define B_   2
#define S_   12
#define N_   512
#define MROWS   (B_*S_*N_)      // 12288 flattened rows for projections
#define SLICES  (H_*B_*S_)      // 192 attention slices (h*B+b)*S+s
#define SLICE_ELEMS (N_*HD_)    // 32768 elements per slice matrix

typedef unsigned short ushort_t;
typedef unsigned int   uint_t;

typedef __attribute__((ext_vector_type(16))) __bf16 v16bf;
typedef __attribute__((ext_vector_type(8)))  float  v8f;

union Frag {
  v16bf    bf;
  ushort_t us[16];
  uint4    q[2];
};

__device__ __forceinline__ ushort_t f2bf(float f) {
  uint_t u = __float_as_uint(f);
  u += 0x7fffu + ((u >> 16) & 1u);   // round-to-nearest-even
  return (ushort_t)(u >> 16);
}

__device__ __forceinline__ v8f wmma_bf16(const Frag& a, const Frag& b, v8f c) {
  return __builtin_amdgcn_wmma_f32_16x16x32_bf16(false, a.bf, false, b.bf,
                                                 (short)0, c, false, false);
}

// fp32 -> bf16, 8 elements/thread, vectorized
__global__ void k_f32_to_bf16(const float* __restrict__ x, ushort_t* __restrict__ y) {
  int idx = (blockIdx.x * blockDim.x + threadIdx.x) * 8;
  float4 f0 = *(const float4*)(x + idx);
  float4 f1 = *(const float4*)(x + idx + 4);
  union { ushort_t u[8]; uint4 q; } o;
  o.u[0] = f2bf(f0.x); o.u[1] = f2bf(f0.y); o.u[2] = f2bf(f0.z); o.u[3] = f2bf(f0.w);
  o.u[4] = f2bf(f1.x); o.u[5] = f2bf(f1.y); o.u[6] = f2bf(f1.z); o.u[7] = f2bf(f1.w);
  *(uint4*)(y + idx) = o.q;
}

// W (512x512 fp32 row-major [k][c]) -> Wt (bf16 col-major [c][k])
__global__ void k_transpose_w(const float* __restrict__ w, ushort_t* __restrict__ wt) {
  int idx = blockIdx.x * blockDim.x + threadIdx.x;   // 0..262143
  int k = idx >> 9;
  int c = idx & 511;
  wt[c * 512 + k] = f2bf(w[k * 512 + c]);
}

// WMMA GEMM: Y(12288x512) = A(bf16 row-major) @ W(bf16 col-major Wt) + bias
// Block: 8 waves, tile 128 rows x 64 cols (4 col tiles per wave).
// mode 0: store bf16 head-split  [slice][n][hd]
// mode 1: store bf16 V-transpose [slice][hd][n]
// mode 2: store fp32 row-major   [r][c]   (final output)
__global__ __launch_bounds__(256) void k_gemm_wmma(
    const ushort_t* __restrict__ A, const ushort_t* __restrict__ Wt,
    const float* __restrict__ bias, void* __restrict__ out, int mode) {
  int wave = threadIdx.x >> 5;
  int lane = threadIdx.x & 31;
  int ln   = lane & 15;
  int rowbase = blockIdx.x * 128 + wave * 16;   // 8 waves -> 128 rows
  int cbase   = blockIdx.y * 64;                // 4 col tiles per wave
  int arow = rowbase + ln;
  int ka = (lane < 16) ? 0 : 8;     // A fragment K base (ISA 16-bit A layout)
  int kb = (lane < 16) ? 0 : 8;     // B fragment K base (ISA 16-bit B layout)
  v8f zero = {};
  v8f acc[4];
#pragma unroll
  for (int t = 0; t < 4; ++t) acc[t] = zero;
  const ushort_t* arp = A + (size_t)arow * 512;
  for (int k0 = 0; k0 < 512; k0 += 32) {
    Frag a;
    a.q[0] = *(const uint4*)(arp + k0 + ka);
    a.q[1] = *(const uint4*)(arp + k0 + ka + 16);
#pragma unroll
    for (int ct = 0; ct < 4; ++ct) {
      const ushort_t* bp = Wt + (size_t)(cbase + ct * 16 + ln) * 512 + k0 + kb;
      __builtin_prefetch(bp + 32, 0, 0);        // next K-step of the B stream
      Frag b;
      const uint4* q = (const uint4*)bp;
      b.q[0] = q[0]; b.q[1] = q[2];
      acc[ct] = wmma_bf16(a, b, acc[ct]);
    }
  }
  int rb2 = rowbase + ((lane >> 4) << 3);   // C/D layout: M = (lane>>4)*8 + r
#pragma unroll
  for (int ct = 0; ct < 4; ++ct) {
    int col = cbase + ct * 16 + ln;
    float bv = bias[col];
    if (mode == 2) {
      float* of = (float*)out;
#pragma unroll
      for (int r = 0; r < 8; ++r)
        of[(size_t)(rb2 + r) * 512 + col] = acc[ct][r] + bv;
    } else {
      ushort_t* ou = (ushort_t*)out;
      int h = col >> 6, hd = col & 63;
#pragma unroll
      for (int r = 0; r < 8; ++r) {
        int row = rb2 + r;
        int bb  = row / (S_ * N_);
        int rem = row - bb * (S_ * N_);
        int ss  = rem >> 9;
        int nn  = rem & 511;
        int slice = (h * B_ + bb) * S_ + ss;
        size_t idx = (size_t)slice * SLICE_ELEMS +
                     (size_t)(mode == 0 ? nn * 64 + hd : hd * 512 + nn);
        ou[idx] = f2bf(acc[ct][r] + bv);
      }
    }
  }
}

// Fused attention: per (slice, 16-row block):
//   scores (g,l,m) via WMMA -> LDS; masking+softmax wave-per-row; P@V via WMMA.
__global__ __launch_bounds__(256) void k_attn(
    const ushort_t* __restrict__ gq, const ushort_t* __restrict__ gk,
    const ushort_t* __restrict__ lq, const ushort_t* __restrict__ lk,
    const ushort_t* __restrict__ mq, const ushort_t* __restrict__ mk,
    const ushort_t* __restrict__ vt, ushort_t* __restrict__ Aout) {
  __shared__ float    sg[16 * 512];
  __shared__ float    sl[16 * 512];
  __shared__ float    sm[16 * 512];
  __shared__ ushort_t P [16 * 512];
  int slice   = blockIdx.y;
  int rowbase = blockIdx.x * 16;
  int wave = threadIdx.x >> 5;
  int lane = threadIdx.x & 31;
  int ln   = lane & 15;
  int ka = (lane < 16) ? 0 : 8;
  int kb = (lane < 16) ? 0 : 8;
  int rloc = (lane >> 4) << 3;
  size_t soff = (size_t)slice * SLICE_ELEMS;
  v8f zero = {};

  // ---- score phase: 3 types x 32 col tiles, 8 waves x 4 tiles each ----
  const ushort_t* Qs[3] = {gq + soff, lq + soff, mq + soff};
  const ushort_t* Ks[3] = {gk + soff, lk + soff, mk + soff};
  float* Ss[3] = {sg, sl, sm};
  for (int ty = 0; ty < 3; ++ty) {
    const ushort_t* Q = Qs[ty];
    const ushort_t* K = Ks[ty];
    v8f acc[4];
#pragma unroll
    for (int t = 0; t < 4; ++t) acc[t] = zero;
#pragma unroll
    for (int k0 = 0; k0 < 64; k0 += 32) {
      Frag a;
      a.q[0] = *(const uint4*)(Q + (rowbase + ln) * 64 + k0 + ka);
      a.q[1] = *(const uint4*)(Q + (rowbase + ln) * 64 + k0 + ka + 16);
#pragma unroll
      for (int t = 0; t < 4; ++t) {
        int jb = (wave * 4 + t) * 16;
        Frag b;
        const uint4* kp = (const uint4*)(K + (jb + ln) * 64 + k0 + kb);
        b.q[0] = kp[0]; b.q[1] = kp[2];
        acc[t] = wmma_bf16(a, b, acc[t]);
      }
    }
    float sc = (ty == 2) ? 1.0f : 0.125f;   // 1/sqrt(HD) for g,l; raw for m
#pragma unroll
    for (int t = 0; t < 4; ++t) {
      int jb = (wave * 4 + t) * 16;
#pragma unroll
      for (int r = 0; r < 8; ++r)
        Ss[ty][(rloc + r) * 512 + jb + ln] = acc[t][r] * sc;
    }
  }
  __syncthreads();

  // ---- row phase: wave handles 2 rows; wave32 shfl reductions ----
  for (int rr = 0; rr < 2; ++rr) {
    int il = wave * 2 + rr;
    int i  = rowbase + il;
    float amax = -3.4e38f; int aidx = 0;
    for (int t = 0; t < 16; ++t) {
      int j = lane + (t << 5);
      float v = sm[il * 512 + j];
      if (v > amax) { amax = v; aidx = j; }
    }
    for (int off = 16; off > 0; off >>= 1) {
      float ov = __shfl_xor(amax, off);
      int   oi = __shfl_xor(aidx, off);
      if (ov > amax || (ov == amax && oi < aidx)) { amax = ov; aidx = oi; }
    }
    float af = (float)aidx;
    // T = sum(relu(att)*|j-a|)/N ; lam_j = (|j-a|*att_j >= T)  (L1 norm cancels)
    float srv = 0.0f;
    for (int t = 0; t < 16; ++t) {
      int j = lane + (t << 5);
      float att = sm[il * 512 + j];
      srv += fmaxf(att, 0.0f) * fabsf((float)j - af);
    }
    for (int off = 16; off > 0; off >>= 1) srv += __shfl_xor(srv, off);
    float T = srv * (1.0f / 512.0f);
    float lg[16];
    float lmax = -3.4e38f;
    for (int t = 0; t < 16; ++t) {
      int j = lane + (t << 5);
      float att = sm[il * 512 + j];
      float g   = sg[il * 512 + j];
      float l   = sl[il * 512 + j];
      bool lam = (fabsf((float)j - af) * att) >= T;
      float v = (j <= i) ? (0.1f * g + (lam ? l : 0.0f)) : -3.4e38f;
      lg[t] = v;
      lmax = fmaxf(lmax, v);
    }
    for (int off = 16; off > 0; off >>= 1) lmax = fmaxf(lmax, __shfl_xor(lmax, off));
    float se = 0.0f;
    for (int t = 0; t < 16; ++t) {
      float e = __expf(lg[t] - lmax);
      lg[t] = e;
      se += e;
    }
    for (int off = 16; off > 0; off >>= 1) se += __shfl_xor(se, off);
    float inv = 1.0f / se;
    for (int t = 0; t < 16; ++t) {
      int j = lane + (t << 5);
      P[il * 512 + j] = f2bf(lg[t] * inv);
    }
  }
  __syncthreads();

  // ---- PV phase: 4 waves, one 16x16 output tile each, K=512 ----
  if (wave < 4) {
    int hdb = wave * 16;
    v8f acc = zero;
    for (int k0 = 0; k0 < 512; k0 += 32) {
      Frag a;
      a.q[0] = *(const uint4*)(&P[ln * 512 + k0 + ka]);
      a.q[1] = *(const uint4*)(&P[ln * 512 + k0 + ka + 16]);
      Frag b;
      const uint4* vp = (const uint4*)(vt + soff + (size_t)(hdb + ln) * 512 + k0 + kb);
      b.q[0] = vp[0]; b.q[1] = vp[2];
      acc = wmma_bf16(a, b, acc);
    }
    int h   = slice / (B_ * S_);
    int rem = slice - h * (B_ * S_);
    int bb  = rem / S_;
    int ss  = rem - bb * S_;
    int col = h * 64 + hdb + ln;
#pragma unroll
    for (int r = 0; r < 8; ++r) {
      int n = rowbase + rloc + r;
      size_t arow = (size_t)((bb * S_ + ss) * N_ + n);
      Aout[arow * 512 + col] = f2bf(acc[r]);
    }
  }
}

extern "C" void kernel_launch(void* const* d_in, const int* in_sizes, int n_in,
                              void* d_out, int out_size, void* d_ws, size_t ws_size,
                              hipStream_t stream) {
  (void)in_sizes; (void)n_in; (void)out_size; (void)ws_size;
  const float* query = (const float*)d_in[0];
  const float* key_  = (const float*)d_in[1];
  const float* value = (const float*)d_in[2];
  const float* w_gq = (const float*)d_in[3];  const float* b_gq = (const float*)d_in[4];
  const float* w_gk = (const float*)d_in[5];  const float* b_gk = (const float*)d_in[6];
  const float* w_gv = (const float*)d_in[7];  const float* b_gv = (const float*)d_in[8];
  const float* w_cq = (const float*)d_in[9];  const float* b_cq = (const float*)d_in[10];
  const float* w_ck = (const float*)d_in[11]; const float* b_ck = (const float*)d_in[12];
  // d_in[13], d_in[14] = w_cv/b_cv : computed but unused in reference output
  const float* w_mq = (const float*)d_in[15]; const float* b_mq = (const float*)d_in[16];
  const float* w_mk = (const float*)d_in[17]; const float* b_mk = (const float*)d_in[18];
  const float* w_o  = (const float*)d_in[19]; const float* b_o  = (const float*)d_in[20];

  // workspace carve-up (~142 MB total)
  char* p = (char*)d_ws;
  const size_t WT_BYTES = (size_t)512 * 512 * 2;
  ushort_t* wt_gq = (ushort_t*)p; p += WT_BYTES;
  ushort_t* wt_gk = (ushort_t*)p; p += WT_BYTES;
  ushort_t* wt_gv = (ushort_t*)p; p += WT_BYTES;
  ushort_t* wt_cq = (ushort_t*)p; p += WT_BYTES;
  ushort_t* wt_ck = (ushort_t*)p; p += WT_BYTES;
  ushort_t* wt_mq = (ushort_t*)p; p += WT_BYTES;
  ushort_t* wt_mk = (ushort_t*)p; p += WT_BYTES;
  ushort_t* wt_o  = (ushort_t*)p; p += WT_BYTES;
  const size_t X_BYTES = (size_t)MROWS * 512 * 2;        // bf16 activations
  ushort_t* xq = (ushort_t*)p; p += X_BYTES;
  ushort_t* xk = (ushort_t*)p; p += X_BYTES;
  ushort_t* xv = (ushort_t*)p; p += X_BYTES;
  const size_t PJ_BYTES = (size_t)SLICES * SLICE_ELEMS * 2;
  ushort_t* pj_gq = (ushort_t*)p; p += PJ_BYTES;
  ushort_t* pj_gk = (ushort_t*)p; p += PJ_BYTES;
  ushort_t* pj_lq = (ushort_t*)p; p += PJ_BYTES;
  ushort_t* pj_lk = (ushort_t*)p; p += PJ_BYTES;
  ushort_t* pj_mq = (ushort_t*)p; p += PJ_BYTES;
  ushort_t* pj_mk = (ushort_t*)p; p += PJ_BYTES;
  ushort_t* vtv   = (ushort_t*)p; p += PJ_BYTES;
  ushort_t* Aout  = (ushort_t*)p; p += PJ_BYTES * 8 / 8; // 12288x512 bf16

  // activation bf16 conversion (6.29M elems each, 8/thread)
  int cgrid = (MROWS * 512) / (256 * 8);
  k_f32_to_bf16<<<cgrid, 256, 0, stream>>>(query, xq);
  k_f32_to_bf16<<<cgrid, 256, 0, stream>>>(key_,  xk);
  k_f32_to_bf16<<<cgrid, 256, 0, stream>>>(value, xv);

  // weight transpose + bf16 convert
  k_transpose_w<<<1024, 256, 0, stream>>>(w_gq, wt_gq);
  k_transpose_w<<<1024, 256, 0, stream>>>(w_gk, wt_gk);
  k_transpose_w<<<1024, 256, 0, stream>>>(w_gv, wt_gv);
  k_transpose_w<<<1024, 256, 0, stream>>>(w_cq, wt_cq);
  k_transpose_w<<<1024, 256, 0, stream>>>(w_ck, wt_ck);
  k_transpose_w<<<1024, 256, 0, stream>>>(w_mq, wt_mq);
  k_transpose_w<<<1024, 256, 0, stream>>>(w_mk, wt_mk);
  k_transpose_w<<<1024, 256, 0, stream>>>(w_o,  wt_o);

  // projections (WMMA), block tile 128x64
  dim3 gg(MROWS / 128, 512 / 64);
  k_gemm_wmma<<<gg, 256, 0, stream>>>(xq, wt_gq, b_gq, pj_gq, 0);
  k_gemm_wmma<<<gg, 256, 0, stream>>>(xk, wt_gk, b_gk, pj_gk, 0);
  k_gemm_wmma<<<gg, 256, 0, stream>>>(xv, wt_gv, b_gv, vtv,   1);
  k_gemm_wmma<<<gg, 256, 0, stream>>>(xq, wt_cq, b_cq, pj_lq, 0);
  k_gemm_wmma<<<gg, 256, 0, stream>>>(xk, wt_ck, b_ck, pj_lk, 0);
  k_gemm_wmma<<<gg, 256, 0, stream>>>(xq, wt_mq, b_mq, pj_mq, 0);
  k_gemm_wmma<<<gg, 256, 0, stream>>>(xk, wt_mk, b_mk, pj_mk, 0);

  // fused masked attention (WMMA scores + masking/softmax + WMMA PV)
  k_attn<<<dim3(N_ / 16, SLICES), 256, 0, stream>>>(pj_gq, pj_gk, pj_lq, pj_lk,
                                                    pj_mq, pj_mk, vtv, Aout);

  // output projection -> d_out (fp32)
  k_gemm_wmma<<<gg, 256, 0, stream>>>(Aout, wt_o, b_o, d_out, 2);
}
